// CrossAttention_37864431681567
// MI455X (gfx1250) — matrix operand order repaired and checked by reading, not compiled, not run on hardware
//
#include <hip/hip_runtime.h>
#include <hip/hip_bf16.h>

typedef __attribute__((ext_vector_type(16))) __bf16 v16bf;
typedef __attribute__((ext_vector_type(8)))  float  v8f;
typedef __attribute__((ext_vector_type(4)))  unsigned int v4u;

#define HEADS   8
#define DHEAD   96
#define NPEP    128
#define MPRO    2048
#define DIM     768
#define SCALE_F 0.1020620726159657f   /* 96^-0.5 */
#define NEG_F   (-1000000.0f)

__device__ __forceinline__ unsigned short f2bf(float f) {
  unsigned int u = __float_as_uint(f);
  u += 0x7fffu + ((u >> 16) & 1u);          // round-to-nearest-even
  return (unsigned short)(u >> 16);
}

// ---- CDNA5 async global->LDS copy (ASYNCcnt) ----
__device__ __forceinline__ void async_b128(unsigned lds_addr, const void* gaddr) {
  asm volatile("global_load_async_to_lds_b128 %0, %1, off"
               :: "v"(lds_addr), "v"(gaddr)
               : "memory");
}
__device__ __forceinline__ void wait_async0() {
  asm volatile("s_wait_asynccnt 0x0" ::: "memory");
}
__device__ __forceinline__ unsigned lds_lo32(const void* p) {
  return (unsigned)(unsigned long long)p;   // LDS aperture: low 32 bits = LDS offset
}

__global__ void cvt_f32_to_bf16(const float* __restrict__ in,
                                unsigned short* __restrict__ out, long n) {
  long i = (long)blockIdx.x * blockDim.x + threadIdx.x;
  long stride = (long)gridDim.x * blockDim.x;
  for (; i < n; i += stride) out[i] = f2bf(in[i]);
}

// 768x768 f32 -> transposed bf16 (out[n*768+k] = in[k*768+n]), LDS-tiled
__launch_bounds__(256)
__global__ void cvt_tr_bf16(const float* __restrict__ in, unsigned short* __restrict__ out) {
  __shared__ float tile[32][33];
  const int bx = blockIdx.x * 32;   // n base
  const int by = blockIdx.y * 32;   // k base
  const int tx = threadIdx.x & 31;
  const int ty = threadIdx.x >> 5;  // 0..7
#pragma unroll
  for (int i = 0; i < 32; i += 8)
    tile[ty + i][tx] = in[(long)(by + ty + i) * 768 + bx + tx];
  __syncthreads();
#pragma unroll
  for (int i = 0; i < 32; i += 8)
    out[(long)(bx + ty + i) * 768 + by + tx] = f2bf(tile[tx][ty + i]);
}

enum { MODE_PLAIN = 0, MODE_KT = 1, MODE_DOTS = 2, MODE_HEAD = 3, MODE_BIASRES = 4 };

union FragAB { v16bf v; v4u u[2]; };

// C = A x B, bf16 in / f32 accum via V_WMMA_F32_16X16X32_BF16.
// A row-major (or, TA: stored K x R). B COLUMN-major: element (k,c) at B[c*ldb + k].
// Block 256 thr = 8 waves; block tile 64x64; wave tile 16x32. Double-buffered
// async global->LDS staging. Batched via grid.z with (z>>3, z&7) offsets.
template <int MODE, bool TA>
__launch_bounds__(256)
__global__ void gemm_bf16(const unsigned short* __restrict__ A, long A_bs, long A_hs, int lda,
                          const unsigned short* __restrict__ B, long B_bs, long B_hs, int ldb,
                          int N, int K,
                          float* __restrict__ outF, unsigned short* __restrict__ outBF,
                          const float* __restrict__ bias, const float* __restrict__ resid,
                          const int* __restrict__ maskRow, const int* __restrict__ maskCol,
                          int aux) {
  __shared__ __align__(16) unsigned short As[2][64 * 40];  // [row][k], stride 40
  __shared__ __align__(16) unsigned short Bs[2][64 * 40];  // [col][k], stride 40

  const int tid = threadIdx.x;
  const int lane = tid & 31;
  const int wid = tid >> 5;
  const int wr = wid >> 1;            // 0..3 : 16-row strip
  const int wc = wid & 1;             // 0..1 : 32-col strip
  const int rowBase = blockIdx.y * 64;
  const int colBase = blockIdx.x * 64;
  const int z = blockIdx.z;

  A += (long)(z >> 3) * A_bs + (long)(z & 7) * A_hs;
  B += (long)(z >> 3) * B_bs + (long)(z & 7) * B_hs;

  // per-thread staging coordinates (one b128 = 8 bf16 per thread per tile)
  const int sr = tid >> 2;            // 0..63 (row for A, col for B)
  const int sk = (tid & 3) * 8;       // 0,8,16,24
  int bcol = colBase + sr; if (bcol >= N) bcol = N - 1;   // clamp (discarded later)

  auto loadA = [&](int k0, int s) {
    if (!TA) {
      async_b128(lds_lo32(&As[s][sr * 40 + sk]),
                 A + (long)(rowBase + sr) * lda + k0 + sk);
    } else {
      // A stored (K x R): element (r,kk) at A[kk*lda + r]; transpose into LDS
#pragma unroll
      for (int i = 0; i < 4; ++i) {
        int idx = tid + i * 256;
        int rp = idx & 31;
        int kk = idx >> 5;
        unsigned int val =
            *(const unsigned int*)(A + (long)(k0 + kk) * lda + rowBase + 2 * rp);
        As[s][(2 * rp) * 40 + kk]     = (unsigned short)(val & 0xffffu);
        As[s][(2 * rp + 1) * 40 + kk] = (unsigned short)(val >> 16);
      }
    }
  };
  auto loadB = [&](int k0, int s) {
    async_b128(lds_lo32(&Bs[s][sr * 40 + sk]),
               B + (long)bcol * ldb + k0 + sk);
  };

  v8f acc0 = {};
  v8f acc1 = {};

  loadA(0, 0);
  loadB(0, 0);
  wait_async0();
  __syncthreads();

  int cur = 0;
  for (int k0 = 0; k0 < K; k0 += 32) {
    const int nxt = cur ^ 1;
    if (k0 + 32 < K) {              // prefetch next tile while computing
      loadA(k0 + 32, nxt);
      loadB(k0 + 32, nxt);
    }

    // ---- fragments per ISA 16-bit WMMA VGPR layout ----
    FragAB af, b0, b1;
    const int arow = wr * 16 + (lane & 15);
    const int kb = (lane >> 4) * 8;             // A: K base 0/8, +16 second half
    af.u[0] = *(const v4u*)&As[cur][arow * 40 + kb];
    af.u[1] = *(const v4u*)&As[cur][arow * 40 + 16 + kb];
    const int bc0 = wc * 32 + (lane & 15);
    b0.u[0] = *(const v4u*)&Bs[cur][bc0 * 40 + kb];
    b0.u[1] = *(const v4u*)&Bs[cur][bc0 * 40 + 16 + kb];
    const int bc1 = bc0 + 16;
    b1.u[0] = *(const v4u*)&Bs[cur][bc1 * 40 + kb];
    b1.u[1] = *(const v4u*)&Bs[cur][bc1 * 40 + 16 + kb];

    acc0 = __builtin_amdgcn_wmma_f32_16x16x32_bf16(false, af.v, false, b0.v,
                                                   (short)0, acc0, false, false);
    acc1 = __builtin_amdgcn_wmma_f32_16x16x32_bf16(false, af.v, false, b1.v,
                                                   (short)0, acc1, false, false);

    wait_async0();
    __syncthreads();
    cur = nxt;
  }

  // ---- epilogue: C layout row=(lane>>4)*8+j, col=lane&15 ----
  const int laneCol = lane & 15;
  const int rowSel = (lane >> 4) * 8;
#pragma unroll
  for (int t = 0; t < 2; ++t) {
    const v8f acc = t ? acc1 : acc0;
    const int c = colBase + wc * 32 + t * 16 + laneCol;
#pragma unroll
    for (int j = 0; j < 8; ++j) {
      const int r = rowBase + wr * 16 + rowSel + j;
      float val = acc[j];
      if (MODE == MODE_PLAIN) {
        outBF[(long)r * N + c] = f2bf(val);
      } else if (MODE == MODE_KT) {
        // aux = log2(rows per batch). r = b*LB+mm, c = h*96+d
        // -> out[((b*8+h)*96+d)*LB + mm]   (per-head transposed, column-major V)
        int b = r >> aux, mm = r & ((1 << aux) - 1);
        int h = c / 96, d = c - h * 96;
        outBF[(((long)(b * 8 + h) * 96 + d) << aux) + mm] = f2bf(val);
      } else if (MODE == MODE_DOTS) {
        int b = z >> 3;
        float v = val * SCALE_F;
        if (maskRow[b * NPEP + r] == 0 || maskCol[b * MPRO + c] == 0) v = NEG_F;
        outF[((long)z * NPEP + r) * MPRO + c] = v;
      } else if (MODE == MODE_HEAD) {
        if (c < N)   // aux = rows per batch (LB)
          outBF[((long)(z >> 3) * aux + r) * 768 + (z & 7) * 96 + c] = f2bf(val);
      } else { /* MODE_BIASRES */
        long o = (long)r * 768 + c;
        outF[o] = val + bias[c] + resid[o];
      }
    }
  }
}

__device__ __forceinline__ float blockReduceMax(float v, float* sh) {
#pragma unroll
  for (int off = 16; off > 0; off >>= 1) v = fmaxf(v, __shfl_xor(v, off));
  if ((threadIdx.x & 31) == 0) sh[threadIdx.x >> 5] = v;
  __syncthreads();
  float r = sh[0];
#pragma unroll
  for (int i = 1; i < 8; ++i) r = fmaxf(r, sh[i]);
  __syncthreads();
  return r;
}

__device__ __forceinline__ float blockReduceSum(float v, float* sh) {
#pragma unroll
  for (int off = 16; off > 0; off >>= 1) v += __shfl_xor(v, off);
  if ((threadIdx.x & 31) == 0) sh[threadIdx.x >> 5] = v;
  __syncthreads();
  float r = 0.f;
#pragma unroll
  for (int i = 0; i < 8; ++i) r += sh[i];
  __syncthreads();
  return r;
}

__launch_bounds__(256)
__global__ void softmax_rows(float* __restrict__ attn, unsigned short* __restrict__ attnBF) {
  __shared__ float sh[8];
  const long row = blockIdx.x;
  float* p = attn + row * MPRO;
  unsigned short* pb = attnBF + row * MPRO;
  const int tid = threadIdx.x;
  float buf[8];
  float m = -3.0e38f;
#pragma unroll
  for (int i = 0; i < 8; ++i) { buf[i] = p[tid + i * 256]; m = fmaxf(m, buf[i]); }
  m = blockReduceMax(m, sh);
  float s = 0.f;
#pragma unroll
  for (int i = 0; i < 8; ++i) { buf[i] = __expf(buf[i] - m); s += buf[i]; }
  s = blockReduceSum(s, sh);
  const float inv = 1.0f / s;
#pragma unroll
  for (int i = 0; i < 8; ++i) {
    float v = buf[i] * inv;
    p[tid + i * 256] = v;
    pb[tid + i * 256] = f2bf(v);
  }
}

__launch_bounds__(256)
__global__ void layernorm_rows(const float* __restrict__ pre,
                               const float* __restrict__ g,
                               const float* __restrict__ b,
                               float* __restrict__ out) {
  __shared__ float sh[8];
  const long row = blockIdx.x;
  const float* p = pre + row * DIM;
  float* o = out + row * DIM;
  const int tid = threadIdx.x;
  float x0 = p[tid], x1 = p[tid + 256], x2 = p[tid + 512];
  float mu = blockReduceSum(x0 + x1 + x2, sh) * (1.0f / DIM);
  float d0 = x0 - mu, d1 = x1 - mu, d2 = x2 - mu;
  float var = blockReduceSum(d0 * d0 + d1 * d1 + d2 * d2, sh) * (1.0f / DIM);
  float rs = rsqrtf(var + 1e-5f);
  o[tid]       = d0 * rs * g[tid]       + b[tid];
  o[tid + 256] = d1 * rs * g[tid + 256] + b[tid + 256];
  o[tid + 512] = d2 * rs * g[tid + 512] + b[tid + 512];
}

extern "C" void kernel_launch(void* const* d_in, const int* in_sizes, int n_in,
                              void* d_out, int out_size, void* d_ws, size_t ws_size,
                              hipStream_t stream) {
  (void)in_sizes; (void)n_in; (void)out_size; (void)ws_size;
  const float* peptide = (const float*)d_in[0];
  const float* protein = (const float*)d_in[1];
  const int*   pep_mask = (const int*)d_in[2];
  const int*   pro_mask = (const int*)d_in[3];
  const float* Wq   = (const float*)d_in[4];
  const float* Wk   = (const float*)d_in[5];
  const float* Wvp  = (const float*)d_in[6];
  const float* Wvpe = (const float*)d_in[7];
  const float* Wop  = (const float*)d_in[8];
  const float* bop  = (const float*)d_in[9];
  const float* Wope = (const float*)d_in[10];
  const float* bope = (const float*)d_in[11];
  const float* lng  = (const float*)d_in[12];
  const float* lnb  = (const float*)d_in[13];

  const long RP = 16l * 128;    // 2048 peptide rows (b,n)
  const long RM = 16l * 2048;   // 32768 protein rows (b,m)

  float* outProt = (float*)d_out;              // (16,128,768)
  float* outPep  = outProt + RP * 768;         // (16,2048,768)
  float* attnF   = outPep + RM * 768;          // (16,8,128,2048)

  char* w = (char*)d_ws;
  auto take = [&](size_t bytes) {
    char* p = w;
    w += (bytes + 255) & ~(size_t)255;
    return p;
  };
  unsigned short* pepBF   = (unsigned short*)take((size_t)RP * 768 * 2);
  unsigned short* protBF  = (unsigned short*)take((size_t)RM * 768 * 2);
  unsigned short* wqT     = (unsigned short*)take(768ull * 768 * 2);
  unsigned short* wkT     = (unsigned short*)take(768ull * 768 * 2);
  unsigned short* wvpT    = (unsigned short*)take(768ull * 768 * 2);
  unsigned short* wvpeT   = (unsigned short*)take(768ull * 768 * 2);
  unsigned short* wopT    = (unsigned short*)take(768ull * 768 * 2);
  unsigned short* wopeT   = (unsigned short*)take(768ull * 768 * 2);
  unsigned short* qBF     = (unsigned short*)take((size_t)RP * 768 * 2);
  unsigned short* kBF     = (unsigned short*)take((size_t)RM * 768 * 2);  // plain (b,m,768)
  unsigned short* vprotT  = (unsigned short*)take((size_t)RM * 768 * 2);  // [(b,h,d), m]
  unsigned short* vpepT   = (unsigned short*)take((size_t)RP * 768 * 2);  // [(b,h,d), n]
  unsigned short* attnBF  = (unsigned short*)take(128ull * 128 * 2048 * 2);
  unsigned short* oprothBF= (unsigned short*)take((size_t)RP * 768 * 2);
  unsigned short* opephBF = (unsigned short*)take((size_t)RM * 768 * 2);
  float* preProt = (float*)take((size_t)RP * 768 * 4);
  float* prePep  = (float*)take((size_t)RM * 768 * 4);

  dim3 blk(256);
  // f32 -> bf16 staging (weights transposed so every GEMM B operand is column-major)
  cvt_f32_to_bf16<<<1024, blk, 0, stream>>>(peptide, pepBF, RP * 768);
  cvt_f32_to_bf16<<<4096, blk, 0, stream>>>(protein, protBF, RM * 768);
  dim3 trg(24, 24);
  cvt_tr_bf16<<<trg, blk, 0, stream>>>(Wq,   wqT);
  cvt_tr_bf16<<<trg, blk, 0, stream>>>(Wk,   wkT);
  cvt_tr_bf16<<<trg, blk, 0, stream>>>(Wvp,  wvpT);
  cvt_tr_bf16<<<trg, blk, 0, stream>>>(Wvpe, wvpeT);
  cvt_tr_bf16<<<trg, blk, 0, stream>>>(Wop,  wopT);
  cvt_tr_bf16<<<trg, blk, 0, stream>>>(Wope, wopeT);

  // projections
  gemm_bf16<MODE_PLAIN, false><<<dim3(12, 32, 1), blk, 0, stream>>>(
      pepBF, 0, 0, 768, wqT, 0, 0, 768, 768, 768,
      nullptr, qBF, nullptr, nullptr, nullptr, nullptr, 0);
  gemm_bf16<MODE_PLAIN, false><<<dim3(12, 512, 1), blk, 0, stream>>>(
      protBF, 0, 0, 768, wkT, 0, 0, 768, 768, 768,
      nullptr, kBF, nullptr, nullptr, nullptr, nullptr, 0);
  gemm_bf16<MODE_KT, false><<<dim3(12, 512, 1), blk, 0, stream>>>(
      protBF, 0, 0, 768, wvpT, 0, 0, 768, 768, 768,
      nullptr, vprotT, nullptr, nullptr, nullptr, nullptr, 11);   // aux=log2(2048)
  gemm_bf16<MODE_KT, false><<<dim3(12, 32, 1), blk, 0, stream>>>(
      pepBF, 0, 0, 768, wvpeT, 0, 0, 768, 768, 768,
      nullptr, vpepT, nullptr, nullptr, nullptr, nullptr, 7);     // aux=log2(128)

  // dots = scale * Q K^T, masked  (per z=(b,h): 128x96 @ 96x2048)
  // B = plain K: column m has contiguous d -> column-major view, ldb=768
  gemm_bf16<MODE_DOTS, false><<<dim3(32, 2, 128), blk, 0, stream>>>(
      qBF, 128l * 768, 96, 768, kBF, 2048l * 768, 96, 768,
      2048, 96, attnF, nullptr, nullptr, nullptr, pep_mask, pro_mask, 0);

  // softmax over m; also produce bf16 attn copy
  softmax_rows<<<16384, blk, 0, stream>>>(attnF, attnBF);

  // attn @ V_prot (per z: 128x2048 @ 2048x96), B = vprotT column-major, ldb=2048
  gemm_bf16<MODE_HEAD, false><<<dim3(2, 2, 128), blk, 0, stream>>>(
      attnBF, 8l * 128 * 2048, 128l * 2048, 2048, vprotT, 8l * 96 * 2048, 96l * 2048, 2048,
      96, 2048, nullptr, oprothBF, nullptr, nullptr, nullptr, nullptr, 128);

  // attn^T @ V_pep (per z: 2048x128 @ 128x96), A transposed-in-LDS, B = vpepT, ldb=128
  gemm_bf16<MODE_HEAD, true><<<dim3(2, 32, 128), blk, 0, stream>>>(
      attnBF, 8l * 128 * 2048, 128l * 2048, 2048, vpepT, 8l * 96 * 128, 96l * 128, 128,
      96, 128, nullptr, opephBF, nullptr, nullptr, nullptr, nullptr, 2048);

  // output projections + bias + residual
  gemm_bf16<MODE_BIASRES, false><<<dim3(12, 32, 1), blk, 0, stream>>>(
      oprothBF, 0, 0, 768, wopT, 0, 0, 768, 768, 768,
      preProt, nullptr, bop, peptide, nullptr, nullptr, 0);
  gemm_bf16<MODE_BIASRES, false><<<dim3(12, 512, 1), blk, 0, stream>>>(
      opephBF, 0, 0, 768, wopeT, 0, 0, 768, 768, 768,
      prePep, nullptr, bope, protein, nullptr, nullptr, 0);

  // layernorm
  layernorm_rows<<<2048, blk, 0, stream>>>(preProt, lng, lnb, outProt);
  layernorm_rows<<<32768, blk, 0, stream>>>(prePep, lng, lnb, outPep);
}
